// Quantize_24429773980080
// MI455X (gfx1250) — matrix operand order repaired and checked by reading, not compiled, not run on hardware
//
#include <hip/hip_runtime.h>

#define VQ_DIM   512
#define VQ_NE    8192
#define VQ_NTOK  16384   // T*B = 512*32
#define VQ_MT    64      // tokens per block in the GEMM
#define VQ_NC    64      // codes per chunk
#define VQ_KT    32      // K tile

typedef __attribute__((ext_vector_type(16))) __bf16  v16bf;
typedef __attribute__((ext_vector_type(8)))  float   v8f;

__device__ __forceinline__ unsigned short f32_to_bf16_rn(float f) {
  unsigned int u = __float_as_uint(f);
  u += 0x7FFFu + ((u >> 16) & 1u);
  return (unsigned short)(u >> 16);
}
__device__ __forceinline__ float bf16_bits_to_f32(unsigned short h) {
  return __uint_as_float(((unsigned int)h) << 16);
}
// Build a 16-element bf16 fragment from two 16B LDS chunks (ds_load_b128 x2).
__device__ __forceinline__ v16bf lds_frag2(const unsigned short* p0,
                                           const unsigned short* p1) {
  union { v16bf v; uint4 q[2]; } u;
  u.q[0] = *(const uint4*)p0;
  u.q[1] = *(const uint4*)p1;
  return u.v;
}
// Raw 32-bit LDS offset of a __shared__ object (generic -> AS(3) -> int).
__device__ __forceinline__ unsigned lds_addr(const void* p) {
  return (unsigned)(unsigned long long)
      (__attribute__((address_space(3))) const void*)p;
}
// CDNA5 async DMA: copy 16B global -> LDS, tracked by ASYNCcnt (ISA 15.18.3).
__device__ __forceinline__ void async_copy16(unsigned lds_off, const void* gptr) {
  asm volatile("global_load_async_to_lds_b128 %0, %1, off"
               :: "v"(lds_off), "v"((unsigned long long)gptr)
               : "memory");
}

// ---------------------------------------------------------------------------
// Kernel 1: per-code squared norms + zero the count histogram.
// ---------------------------------------------------------------------------
__global__ __launch_bounds__(256) void vq_prep(const float* __restrict__ embed,
                                               float* __restrict__ esq,
                                               int* __restrict__ counts) {
  int j = blockIdx.x * 256 + threadIdx.x;          // 0..VQ_NE-1
  float s = 0.0f;
  for (int k = 0; k < VQ_DIM; ++k) {               // coalesced across j
    float e = embed[(size_t)k * VQ_NE + j];
    s += e * e;
  }
  esq[j] = s;
  counts[j] = 0;
}

// ---------------------------------------------------------------------------
// Kernel 2: transpose codebook [DIM][NE] -> [NE][DIM] (f32 for the gather)
// and emit bf16 hi/lo split of the transposed codebook for the WMMA GEMM.
// ---------------------------------------------------------------------------
__global__ __launch_bounds__(256) void vq_transpose(
    const float* __restrict__ embed, float* __restrict__ embedT,
    unsigned short* __restrict__ Bhi_g, unsigned short* __restrict__ Blo_g) {
  __shared__ float tile[32][33];
  int jb = blockIdx.x * 32, kb = blockIdx.y * 32;
  int tx = threadIdx.x, ty = threadIdx.y;          // 32 x 8
  #pragma unroll
  for (int s = 0; s < 32; s += 8)
    tile[ty + s][tx] = embed[(size_t)(kb + ty + s) * VQ_NE + jb + tx];
  __syncthreads();
  #pragma unroll
  for (int s = 0; s < 32; s += 8) {
    size_t o = (size_t)(jb + ty + s) * VQ_DIM + kb + tx;
    float x = tile[tx][ty + s];
    unsigned short h = f32_to_bf16_rn(x);
    embedT[o] = x;
    Bhi_g[o] = h;
    Blo_g[o] = f32_to_bf16_rn(x - bf16_bits_to_f32(h));
  }
}

// ---------------------------------------------------------------------------
// Kernel 3: bf16 hi/lo split of the input tokens (done ONCE, not per chunk).
// ---------------------------------------------------------------------------
__global__ __launch_bounds__(256) void vq_convert_input(
    const float* __restrict__ input,
    unsigned short* __restrict__ Ahi_g, unsigned short* __restrict__ Alo_g) {
  size_t base = (size_t)blockIdx.x * 2048 + threadIdx.x;
  #pragma unroll
  for (int s = 0; s < 8; ++s) {
    size_t i = base + (size_t)s * 256;
    float x = input[i];
    unsigned short h = f32_to_bf16_rn(x);
    Ahi_g[i] = h;
    Alo_g[i] = f32_to_bf16_rn(x - bf16_bits_to_f32(h));
  }
}

// ---------------------------------------------------------------------------
// Kernel 4: fused distance GEMM (bf16x3 via v_wmma_f32_16x16x32_bf16) + argmin.
// Block = 256 threads = 8 waves (wave32), arranged 4(M) x 2(N).
// A slab (64 tokens x 512 K, hi+lo) persistent in LDS, dense XOR-swizzled
// (chunk = row*64 + (seg ^ (row&15)) -> conflict-free b128, no padding).
// B tiles triple-buffered async DMA -> single barrier per K-step.
// ---------------------------------------------------------------------------
__global__ __launch_bounds__(256) void vq_gemm_argmin(
    const unsigned short* __restrict__ Ahi_g,  // [NTOK][DIM] bf16-hi
    const unsigned short* __restrict__ Alo_g,  // [NTOK][DIM] bf16-lo
    const unsigned short* __restrict__ Bhi_g,  // [NE][DIM]  bf16-hi (transposed)
    const unsigned short* __restrict__ Blo_g,  // [NE][DIM]  bf16-lo
    const float* __restrict__ esq,             // [NE]
    int* __restrict__ out_idx)                 // [NTOK]
{
  __shared__ __align__(16) unsigned short AhiL[VQ_MT * VQ_DIM];   // 64 KB
  __shared__ __align__(16) unsigned short AloL[VQ_MT * VQ_DIM];   // 64 KB
  // B: stride 40 ushorts = 80B -> 16B aligned, conflict-free (20%64); 3 bufs.
  __shared__ __align__(16) unsigned short BhiL[3][VQ_NC][40];
  __shared__ __align__(16) unsigned short BloL[3][VQ_NC][40];
  __shared__ float redS[2][VQ_MT];
  __shared__ int   redI[2][VQ_MT];

  const int tid   = threadIdx.x;
  const int lane  = tid & 31;
  const int wave  = tid >> 5;
  const int waveM = wave & 3;          // 0..3 -> 16-row slab
  const int waveN = wave >> 2;         // 0..1 -> 32-col slab
  const int r     = lane & 15;
  const int half  = lane >> 4;         // 0 or 1
  const int mbase = blockIdx.x * VQ_MT;

  // CDNA5 16-bit A layout: lanes 0-15 hold K{0..7,16..23}, lanes 16-31 K{8..15,24..31}
  // CDNA5 16-bit B layout: lanes 0-15 hold K{0..7,16..23}, lanes 16-31 K{8..15,24..31}
  const int koffB = half * 8;
  const int arow  = waveM * 16 + r;

  // ---- prologue: async-stage the persistent A slab (64 x 512, hi+lo) ----
  #pragma unroll
  for (int s = 0; s < 16; ++s) {
    int seg = tid + s * 256;                   // 0..4095 (16B segments)
    int row = seg >> 6, s8 = seg & 63;         // 64 segs per 512-ushort row
    int chunk = row * 64 + (s8 ^ (row & 15));  // XOR swizzle
    const size_t go = (size_t)(mbase + row) * VQ_DIM + s8 * 8;
    async_copy16(lds_addr(&AhiL[chunk * 8]), Ahi_g + go);
    async_copy16(lds_addr(&AloL[chunk * 8]), Alo_g + go);
  }
  // B tile stage: 64 cols x 32 K, hi+lo -> 2 async b128 per thread.
  auto stage_B = [&](int jbX, int ktX, int b) {
    int col = tid >> 2, sk = (tid & 3) * 8;
    const size_t go = (size_t)(jbX + col) * VQ_DIM + ktX + sk;
    async_copy16(lds_addr(&BhiL[b][col][sk]), Bhi_g + go);
    async_copy16(lds_addr(&BloL[b][col][sk]), Blo_g + go);
  };
  stage_B(0, 0, 0);

  float bestS[2][8];
  int   bestI[2][8];
  #pragma unroll
  for (int ns = 0; ns < 2; ++ns)
    #pragma unroll
    for (int v = 0; v < 8; ++v) { bestS[ns][v] = 3.4e38f; bestI[ns][v] = 0; }

  int bcur = 0;
  #pragma unroll 1
  for (int jb = 0; jb < VQ_NE; jb += VQ_NC) {
    v8f acc[2] = {};
    #pragma unroll 1
    for (int kt = 0; kt < VQ_DIM; kt += VQ_KT) {
      const int bnext = (bcur == 2) ? 0 : bcur + 1;
      int kt_n = kt + VQ_KT, jb_n = jb;
      if (kt_n == VQ_DIM) { kt_n = 0; jb_n += VQ_NC; }  // final step: harmless
      stage_B(jb_n, kt_n, bnext);                       // dummy stage into ws
      // newest 2 copies (next tile) may be in flight; current tile is done.
      asm volatile("s_wait_asynccnt 0x2" ::: "memory");
      __syncthreads();   // single barrier/step (safe with triple buffering)

      const int ks = (kt >> 3) + half;          // 16B segment index within row
      const int c0 = arow * 64 + (ks ^ r);
      const int c1 = arow * 64 + ((ks + 2) ^ r);
      v16bf a_hi = lds_frag2(AhiL + c0 * 8, AhiL + c1 * 8);
      v16bf a_lo = lds_frag2(AloL + c0 * 8, AloL + c1 * 8);
      #pragma unroll
      for (int ns = 0; ns < 2; ++ns) {
        int bcol = waveN * 32 + ns * 16 + r;
        v16bf b_hi = lds_frag2(&BhiL[bcur][bcol][koffB], &BhiL[bcur][bcol][koffB + 16]);
        v16bf b_lo = lds_frag2(&BloL[bcur][bcol][koffB], &BloL[bcur][bcol][koffB + 16]);
        // bf16x3: hi*hi + hi*lo + lo*hi, f32 accumulate
        acc[ns] = __builtin_amdgcn_wmma_f32_16x16x32_bf16(
            false, a_hi, false, b_hi, (short)0, acc[ns], false, false);
        acc[ns] = __builtin_amdgcn_wmma_f32_16x16x32_bf16(
            false, a_hi, false, b_lo, (short)0, acc[ns], false, false);
        acc[ns] = __builtin_amdgcn_wmma_f32_16x16x32_bf16(
            false, a_lo, false, b_hi, (short)0, acc[ns], false, false);
      }
      bcur = bnext;
    }
    // ---- score = ||e||^2 - 2*x.e ; update per-lane running argmin ----
    #pragma unroll
    for (int ns = 0; ns < 2; ++ns) {
      int j = jb + waveN * 32 + ns * 16 + r;   // C layout: N = lane&15
      float e2 = esq[j];
      #pragma unroll
      for (int v = 0; v < 8; ++v) {
        float sc = e2 - 2.0f * acc[ns][v];
        if (sc < bestS[ns][v]) { bestS[ns][v] = sc; bestI[ns][v] = j; }
      }
    }
  }

  // ---- reduce: combine nsub in-lane, then across 16 lanes per row ----
  #pragma unroll
  for (int v = 0; v < 8; ++v) {
    float s0 = bestS[0][v]; int i0 = bestI[0][v];
    float s1 = bestS[1][v]; int i1 = bestI[1][v];
    if (s1 < s0 || (s1 == s0 && i1 < i0)) { s0 = s1; i0 = i1; }
    #pragma unroll
    for (int off = 8; off >= 1; off >>= 1) {
      float so = __shfl_xor(s0, off, 32);
      int   io = __shfl_xor(i0, off, 32);
      if (so < s0 || (so == s0 && io < i0)) { s0 = so; i0 = io; }
    }
    // C layout: lanes 0-15 -> row v, lanes 16-31 -> row v+8
    if (r == 0) {
      int row_local = waveM * 16 + v + half * 8;
      redS[waveN][row_local] = s0;
      redI[waveN][row_local] = i0;
    }
  }
  __syncthreads();
  if (tid < VQ_MT) {   // combine the two column-slab waves (ties -> lowest idx)
    float s0 = redS[0][tid]; int i0 = redI[0][tid];
    float s1 = redS[1][tid]; int i1 = redI[1][tid];
    if (s1 < s0 || (s1 == s0 && i1 < i0)) { s0 = s1; i0 = i1; }
    out_idx[mbase + tid] = i0;
  }
}

// ---------------------------------------------------------------------------
// Kernel 5: gather + mask + per-block diff partials + count histogram.
// ---------------------------------------------------------------------------
__global__ __launch_bounds__(256) void vq_gather(
    const float* __restrict__ input, const int* __restrict__ mask,
    const float* __restrict__ embedT, const int* __restrict__ idx,
    float* __restrict__ q_out, float* __restrict__ ind_out,
    int* __restrict__ counts, float* __restrict__ partial)
{
  __shared__ float red[256];
  const int tok = blockIdx.x * 2 + (threadIdx.x >> 7);
  const int d4  = threadIdx.x & 127;
  const int id  = idx[tok];
  const int mi  = mask[tok];
  const float m = mi ? 1.0f : 0.0f;

  const float4 e = ((const float4*)(embedT + (size_t)id  * VQ_DIM))[d4];
  const float4 x = ((const float4*)(input  + (size_t)tok * VQ_DIM))[d4];
  float4 q;
  q.x = e.x * m; q.y = e.y * m; q.z = e.z * m; q.w = e.w * m;
  ((float4*)q_out)[(size_t)tok * (VQ_DIM / 4) + d4] = q;

  float dx = e.x - x.x, dy = e.y - x.y, dz = e.z - x.z, dw = e.w - x.w;
  float dsum = m * (dx * dx + dy * dy + dz * dz + dw * dw);

  if (d4 == 0) {
    ind_out[tok] = (float)id;
    if (mi) atomicAdd(&counts[id], 1);   // integer atomic: deterministic
  }
  red[threadIdx.x] = dsum;
  __syncthreads();
  #pragma unroll
  for (int s = 128; s >= 1; s >>= 1) {
    if (threadIdx.x < s) red[threadIdx.x] += red[threadIdx.x + s];
    __syncthreads();
  }
  if (threadIdx.x == 0) partial[blockIdx.x] = red[0];
}

// ---------------------------------------------------------------------------
// Kernel 6: final reductions (fixed order -> deterministic).
// ---------------------------------------------------------------------------
__global__ __launch_bounds__(256) void vq_finalize(
    const float* __restrict__ partial, int nPartial,
    const int* __restrict__ counts,
    float* __restrict__ diff_out, float* __restrict__ eff_out)
{
  __shared__ float sh[256];
  const int t = threadIdx.x;

  float s = 0.0f;
  for (int i = t; i < nPartial; i += 256) s += partial[i];
  sh[t] = s; __syncthreads();
  for (int st = 128; st >= 1; st >>= 1) {
    if (t < st) sh[t] += sh[t + st];
    __syncthreads();
  }
  if (t == 0) *diff_out = sh[0] / (float)((size_t)VQ_NTOK * VQ_DIM);
  __syncthreads();

  float cs = 0.0f, c2 = 0.0f;
  for (int i = t; i < VQ_NE; i += 256) {
    float c = (float)counts[i];
    cs += c; c2 += c * c;
  }
  sh[t] = cs; __syncthreads();
  for (int st = 128; st >= 1; st >>= 1) {
    if (t < st) sh[t] += sh[t + st];
    __syncthreads();
  }
  float totc = sh[0]; __syncthreads();
  sh[t] = c2; __syncthreads();
  for (int st = 128; st >= 1; st >>= 1) {
    if (t < st) sh[t] += sh[t + st];
    __syncthreads();
  }
  if (t == 0) *eff_out = (sh[0] > 0.0f) ? (totc * totc / sh[0]) : 0.0f;
}

// ---------------------------------------------------------------------------
extern "C" void kernel_launch(void* const* d_in, const int* in_sizes, int n_in,
                              void* d_out, int out_size, void* d_ws, size_t ws_size,
                              hipStream_t stream) {
  const float* input = (const float*)d_in[0];   // [T,B,DIM] f32
  const int*   mask  = (const int*)d_in[1];     // [T,B] bool->int
  const float* embed = (const float*)d_in[2];   // [DIM,NE] f32
  float* out = (float*)d_out;

  // workspace layout
  char* ws = (char*)d_ws;
  size_t off = 0;
  float* embedT = (float*)(ws + off); off += (size_t)VQ_NE * VQ_DIM * sizeof(float);
  unsigned short* Ahi_g = (unsigned short*)(ws + off); off += (size_t)VQ_NTOK * VQ_DIM * 2;
  unsigned short* Alo_g = (unsigned short*)(ws + off); off += (size_t)VQ_NTOK * VQ_DIM * 2;
  unsigned short* Bhi_g = (unsigned short*)(ws + off); off += (size_t)VQ_NE * VQ_DIM * 2;
  unsigned short* Blo_g = (unsigned short*)(ws + off); off += (size_t)VQ_NE * VQ_DIM * 2;
  float* esq     = (float*)(ws + off); off += (size_t)VQ_NE * sizeof(float);
  int*   counts  = (int*)  (ws + off); off += (size_t)VQ_NE * sizeof(int);
  int*   idx     = (int*)  (ws + off); off += (size_t)VQ_NTOK * sizeof(int);
  float* partial = (float*)(ws + off);

  // output layout: quantize_st | diff | embed_ind | effective_units
  float* q_out    = out;
  float* diff_out = out + (size_t)VQ_NTOK * VQ_DIM;
  float* ind_out  = diff_out + 1;
  float* eff_out  = ind_out + VQ_NTOK;

  vq_prep<<<VQ_NE / 256, 256, 0, stream>>>(embed, esq, counts);
  vq_transpose<<<dim3(VQ_NE / 32, VQ_DIM / 32), dim3(32, 8), 0, stream>>>(
      embed, embedT, Bhi_g, Blo_g);
  vq_convert_input<<<(VQ_NTOK * VQ_DIM) / 2048, 256, 0, stream>>>(input, Ahi_g, Alo_g);
  vq_gemm_argmin<<<VQ_NTOK / VQ_MT, 256, 0, stream>>>(Ahi_g, Alo_g, Bhi_g, Blo_g,
                                                      esq, idx);
  vq_gather<<<VQ_NTOK / 2, 256, 0, stream>>>(input, mask, embedT, idx,
                                             q_out, ind_out, counts, partial);
  vq_finalize<<<1, 256, 0, stream>>>(partial, VQ_NTOK / 2, counts, diff_out, eff_out);
}
